// GCP_BIN_CNN_16123307229940
// MI455X (gfx1250) — matrix operand order repaired and checked by reading, not compiled, not run on hardware
//
#include <hip/hip_runtime.h>
#include <stdint.h>

#define HID 128

typedef __attribute__((ext_vector_type(16))) __bf16 v16bf;
typedef __attribute__((ext_vector_type(8)))  float  v8f;

union Frag { v16bf bf; uint32_t u[8]; };

__device__ __forceinline__ unsigned short f2bf(float f) {
  union { float f; uint32_t u; } x; x.f = f;
  uint32_t u = x.u;
  return (unsigned short)((u + 0x7FFFu + ((u >> 16) & 1u)) >> 16);
}

__device__ __forceinline__ float sigf(float x) { return 1.0f / (1.0f + __expf(-x)); }

// ---- WMMA fragment loaders (16x16x32 bf16, wave32; contiguous k = 16*(lane/16)+s, same order for A and B) ----
// A 16x32: lanes 0-15 row M=lane, K=0..15 in v0..7 (packed pairs);
//          lanes 16-31 same rows, K=16..31.
__device__ __forceinline__ v16bf load_frag_A(const uint32_t* s, int mrow, int ld2, int kk) {
  int lane = threadIdx.x & 31;
  Frag a;
  const uint32_t* p = s + (mrow + (lane & 15)) * ld2 + (kk >> 1) + ((lane >> 4) << 3);
#pragma unroll
  for (int v = 0; v < 4; ++v) a.u[v] = p[v];
#pragma unroll
  for (int v = 0; v < 4; ++v) a.u[4 + v] = p[4 + v];
  return a.bf;
}

// B 32x16 where B[k][n] = W[n][k], W row-major [N][K] bf16 (k pairs contiguous):
// lanes 0-15: N=lane, K=0..15 in v0..7; lanes 16-31: N=lane-16, K=16..31.
__device__ __forceinline__ v16bf load_frag_B(const uint32_t* w, int ncol, int ld2, int kk) {
  int lane = threadIdx.x & 31;
  Frag b;
  const uint32_t* p = w + (ncol + (lane & 15)) * ld2 + (kk >> 1) + ((lane >> 4) << 3);
#pragma unroll
  for (int v = 0; v < 8; ++v) b.u[v] = p[v];
  return b.bf;
}

__device__ __forceinline__ v8f wmma_bf16(v16bf a, v16bf b, v8f c) {
  return __builtin_amdgcn_wmma_f32_16x16x32_bf16(false, a, false, b, (short)0, c, false, false);
}

// One wave computes a 16(M) x 64(N) tile: bias + optional relu, bf16 store to LDS.
template <int KD, int LDIN, bool RELU>
__device__ __forceinline__ void wave_tile(const uint32_t* s_in, unsigned short* s_out,
                                          int ostride, const unsigned short* W,
                                          const float* bias, int mrow, int nbase) {
  v8f acc[4];
#pragma unroll
  for (int j = 0; j < 4; ++j)
#pragma unroll
    for (int r = 0; r < 8; ++r) acc[j][r] = 0.0f;
  const uint32_t* w32 = (const uint32_t*)W;
#pragma unroll
  for (int kk = 0; kk < KD; kk += 32) {
    v16bf a = load_frag_A(s_in, mrow, LDIN, kk);
#pragma unroll
    for (int j = 0; j < 4; ++j) {
      v16bf b = load_frag_B(w32, nbase + 16 * j, KD / 2, kk);
      acc[j] = wmma_bf16(a, b, acc[j]);
    }
  }
  int lane = threadIdx.x & 31;
  int nlo  = lane & 15;
  int mh   = (lane >> 4) * 8;
#pragma unroll
  for (int j = 0; j < 4; ++j) {
    int cn   = nbase + 16 * j + nlo;
    float bv = bias[cn];
#pragma unroll
    for (int r = 0; r < 8; ++r) {
      float v = acc[j][r] + bv;
      if (RELU) v = fmaxf(v, 0.0f);
      s_out[(mrow + mh + r) * ostride + cn] = f2bf(v);
    }
  }
}

// ---- small prep kernels ----
__global__ void f32_to_bf16_kernel(const float* __restrict__ s, unsigned short* __restrict__ d, int n) {
  int i = blockIdx.x * 256 + threadIdx.x;
  if (i < n) d[i] = f2bf(s[i]);
}

__global__ void build_wc_kernel(const float* __restrict__ wih, const float* __restrict__ whh,
                                unsigned short* __restrict__ wc) {
  int i = blockIdx.x * 256 + threadIdx.x;  // 512*512
  int g = i >> 9, k = i & 511;
  float v = (k < 384) ? wih[g * 384 + k] : whh[g * 128 + (k - 384)];
  wc[i] = f2bf(v);
}

__global__ void gather_embed_kernel(const int* __restrict__ q, const float* __restrict__ embed,
                                    unsigned short* __restrict__ xb, unsigned short* __restrict__ hb,
                                    int total) {
  int i = blockIdx.x * 256 + threadIdx.x;
  if (i >= total) return;
  int cell = i >> 7, d = i & 127;
  unsigned short v = f2bf(embed[q[cell] * HID + d]);
  xb[i] = v;
  hb[i] = v;
}

// ---- edge MLP + scatter-add (segment_sum) ----
#define LPA 132  // 256 bf16 cols = 128 dwords, +4 pad dwords
#define LPB 68   // 128 bf16 cols = 64 dwords, +4 pad dwords

__global__ __launch_bounds__(256) void edge_mlp_kernel(
    const unsigned short* __restrict__ h,
    const int* __restrict__ esrc, const int* __restrict__ edst,
    const unsigned short* __restrict__ W0, const float* __restrict__ b0,
    const unsigned short* __restrict__ W1, const float* __restrict__ b1,
    const unsigned short* __restrict__ W2, const float* __restrict__ b2,
    const unsigned short* __restrict__ W3, const float* __restrict__ b3,
    float* __restrict__ mout, int nE) {
  __shared__ uint32_t s_a[64 * LPA];  // 33 KB: cat(h[src],h[dst]) bf16 / layer ping
  __shared__ uint32_t s_b[64 * LPB];  // 17 KB: layer pong
  const uint32_t* h32 = (const uint32_t*)h;
  int t = threadIdx.x;
  int ebase = blockIdx.x * 64;

  // gather cat(h[src], h[dst]) -> LDS (zero-pad tail edges; keeps EXEC all-1s for WMMA)
#pragma unroll
  for (int i = 0; i < 32; ++i) {
    int dw = t + i * 256;  // 0..8191 logical dwords of [64][128]
    int el = dw >> 7, k2 = dw & 127;
    int e = ebase + el;
    uint32_t val = 0u;
    if (e < nE) {
      int cell = (k2 < 64) ? esrc[e] : edst[e];
      val = h32[cell * 64 + (k2 & 63)];
    }
    s_a[el * LPA + k2] = val;
  }
  __syncthreads();

  int wave = t >> 5;
  int mrow = (wave & 3) * 16;   // 4 M-tiles of 16 edges
  int nbase = (wave >> 2) * 64; // 2 N-halves of 64 outputs

  wave_tile<256, LPA, true>(s_a, (unsigned short*)s_b, 2 * LPB, W0, b0, mrow, nbase);
  __syncthreads();
  wave_tile<128, LPB, true>(s_b, (unsigned short*)s_a, 2 * LPA, W1, b1, mrow, nbase);
  __syncthreads();
  wave_tile<128, LPA, true>(s_a, (unsigned short*)s_b, 2 * LPB, W2, b2, mrow, nbase);
  __syncthreads();

  // final layer: no relu; scatter-add into message buffer (segment_sum over dst)
  {
    v8f acc[4];
#pragma unroll
    for (int j = 0; j < 4; ++j)
#pragma unroll
      for (int r = 0; r < 8; ++r) acc[j][r] = 0.0f;
    const uint32_t* w32 = (const uint32_t*)W3;
#pragma unroll
    for (int kk = 0; kk < 128; kk += 32) {
      v16bf a = load_frag_A(s_b, mrow, LPB, kk);
#pragma unroll
      for (int j = 0; j < 4; ++j) {
        v16bf b = load_frag_B(w32, nbase + 16 * j, 64, kk);
        acc[j] = wmma_bf16(a, b, acc[j]);
      }
    }
    int lane = t & 31;
    int nlo  = lane & 15;
    int mh   = (lane >> 4) * 8;
#pragma unroll
    for (int r = 0; r < 8; ++r) {
      int m = mrow + mh + r;
      int e = ebase + m;
      if (e < nE) {
        int cell = edst[e];
#pragma unroll
        for (int j = 0; j < 4; ++j) {
          int cn = nbase + 16 * j + nlo;
          atomicAdd(&mout[(size_t)cell * HID + cn], acc[j][r] + b3[cn]);
        }
      }
    }
  }
}

// ---- fused LSTM cell + score ----
#define LPG 260  // 512 bf16 cols = 256 dwords, +4 pad
#define LGG 516  // gates row stride (f32), padded

__global__ __launch_bounds__(256) void lstm_kernel(
    const unsigned short* __restrict__ xb,
    const unsigned short* rhb,                 // aliases rh_out: no restrict
    const float* __restrict__ mi, const float* __restrict__ mo,
    const unsigned short* __restrict__ Wc, const float* __restrict__ sw,
    unsigned short* __restrict__ hb, unsigned short* rh_out,
    float* __restrict__ rc, float* __restrict__ logits) {
  __shared__ uint32_t s_a[16 * LPG];  // 16.6 KB: cat(x, m_intra, m_inter, rh) bf16
  __shared__ float s_g[16 * LGG];     // 33 KB: gates f32
  __shared__ float s_red[256];
  int t = threadIdx.x;
  int cell0 = blockIdx.x * 16;
  const uint32_t* x32  = (const uint32_t*)xb;
  const uint32_t* rh32 = (const uint32_t*)rhb;

#pragma unroll
  for (int i = 0; i < 16; ++i) {
    int dw = t + i * 256;  // [16][256] logical dwords
    int row = dw >> 8, k2 = dw & 255;
    int cell = cell0 + row;
    int k = k2 << 1;
    uint32_t val;
    if (k < 128) {
      val = x32[cell * 64 + k2];
    } else if (k < 256) {
      float a = mi[(size_t)cell * HID + (k - 128)];
      float b = mi[(size_t)cell * HID + (k - 127)];
      val = (uint32_t)f2bf(a) | ((uint32_t)f2bf(b) << 16);
    } else if (k < 384) {
      float a = mo[(size_t)cell * HID + (k - 256)];
      float b = mo[(size_t)cell * HID + (k - 255)];
      val = (uint32_t)f2bf(a) | ((uint32_t)f2bf(b) << 16);
    } else {
      val = rh32[cell * 64 + (k2 - 192)];
    }
    s_a[row * LPG + k2] = val;
  }
  __syncthreads();

  int wave = t >> 5, lane = t & 31;
  int nbase = wave * 64;  // 8 waves x 64 gate columns = 512
  v8f acc[4];
#pragma unroll
  for (int j = 0; j < 4; ++j)
#pragma unroll
    for (int r = 0; r < 8; ++r) acc[j][r] = 0.0f;
  const uint32_t* w32 = (const uint32_t*)Wc;
#pragma unroll 4
  for (int kk = 0; kk < 512; kk += 32) {
    v16bf a = load_frag_A(s_a, 0, LPG, kk);
#pragma unroll
    for (int j = 0; j < 4; ++j) {
      v16bf b = load_frag_B(w32, nbase + 16 * j, 256, kk);
      acc[j] = wmma_bf16(a, b, acc[j]);
    }
  }
  int nlo = lane & 15, mh = (lane >> 4) * 8;
#pragma unroll
  for (int j = 0; j < 4; ++j)
#pragma unroll
    for (int r = 0; r < 8; ++r)
      s_g[(mh + r) * LGG + nbase + 16 * j + nlo] = acc[j][r];
  __syncthreads();

  // fused cell update + score partial
  int c = t >> 4;
  int dbase = (t & 15) * 8;
  int cell = cell0 + c;
  const float* g = &s_g[c * LGG];
  float part = 0.0f;
#pragma unroll
  for (int dd = 0; dd < 8; ++dd) {
    int d = dbase + dd;
    float iv = g[d], fv = g[128 + d], gv = g[256 + d], ov = g[384 + d];
    float rcold = rc[(size_t)cell * HID + d];
    float ncv = sigf(fv) * rcold + sigf(iv) * tanhf(gv);
    float nhv = sigf(ov) * tanhf(ncv);
    rc[(size_t)cell * HID + d] = ncv;
    unsigned short hv = f2bf(nhv);
    hb[(size_t)cell * HID + d] = hv;
    rh_out[(size_t)cell * HID + d] = hv;
    part += nhv * sw[d];
  }
  s_red[t] = part;
  __syncthreads();
  if ((t & 15) == 0) {
    float s = 0.0f;
#pragma unroll
    for (int i = 0; i < 16; ++i) s += s_red[(c << 4) + i];
    logits[cell] = s;
  }
}

extern "C" void kernel_launch(void* const* d_in, const int* in_sizes, int n_in,
                              void* d_out, int out_size, void* d_ws, size_t ws_size,
                              hipStream_t stream) {
  const int*   q    = (const int*)d_in[0];
  const int*   esrc = (const int*)d_in[1];
  const int*   edst = (const int*)d_in[2];
  const float* embed = (const float*)d_in[3];
  const float* W0f = (const float*)d_in[4];
  const float* b0  = (const float*)d_in[5];
  const float* W1f = (const float*)d_in[6];
  const float* b1  = (const float*)d_in[7];
  const float* W2f = (const float*)d_in[8];
  const float* b2  = (const float*)d_in[9];
  const float* W3f = (const float*)d_in[10];
  const float* b3  = (const float*)d_in[11];
  const float* Wih = (const float*)d_in[12];
  const float* Whh = (const float*)d_in[13];
  const float* sw  = (const float*)d_in[14];

  const int cells = in_sizes[0];      // 98304
  const int E     = in_sizes[1] / 2;  // 250000

  char* ws = (char*)d_ws;
  size_t off = 0;
  auto alloc = [&](size_t bytes) -> char* {
    char* p = ws + off;
    off += (bytes + 255) & ~(size_t)255;
    return p;
  };
  unsigned short* xb  = (unsigned short*)alloc((size_t)cells * HID * 2);
  unsigned short* hb  = (unsigned short*)alloc((size_t)cells * HID * 2);
  unsigned short* rhb = (unsigned short*)alloc((size_t)cells * HID * 2);
  float* rc = (float*)alloc((size_t)cells * HID * 4);
  float* mi = (float*)alloc((size_t)cells * HID * 4);
  float* mo = (float*)alloc((size_t)cells * HID * 4);
  unsigned short* W0b = (unsigned short*)alloc((size_t)2 * 128 * 256 * 2);
  unsigned short* W1b = (unsigned short*)alloc((size_t)2 * 128 * 128 * 2);
  unsigned short* W2b = (unsigned short*)alloc((size_t)2 * 128 * 128 * 2);
  unsigned short* W3b = (unsigned short*)alloc((size_t)2 * 128 * 128 * 2);
  unsigned short* Wcb = (unsigned short*)alloc((size_t)512 * 512 * 2);

  // bf16 weights
  f32_to_bf16_kernel<<<(2 * 128 * 256 + 255) / 256, 256, 0, stream>>>(W0f, W0b, 2 * 128 * 256);
  f32_to_bf16_kernel<<<(2 * 128 * 128 + 255) / 256, 256, 0, stream>>>(W1f, W1b, 2 * 128 * 128);
  f32_to_bf16_kernel<<<(2 * 128 * 128 + 255) / 256, 256, 0, stream>>>(W2f, W2b, 2 * 128 * 128);
  f32_to_bf16_kernel<<<(2 * 128 * 128 + 255) / 256, 256, 0, stream>>>(W3f, W3b, 2 * 128 * 128);
  build_wc_kernel<<<(512 * 512) / 256, 256, 0, stream>>>(Wih, Whh, Wcb);

  // state init: x = h = embed[q] (bf16), rh = 0, rc = 0
  gather_embed_kernel<<<((size_t)cells * HID + 255) / 256, 256, 0, stream>>>(q, embed, xb, hb,
                                                                             cells * HID);
  hipMemsetAsync(rhb, 0, (size_t)cells * HID * 2, stream);
  hipMemsetAsync(rc, 0, (size_t)cells * HID * 4, stream);

  const int eblocks = (E + 63) / 64;
  const int lblocks = cells / 16;
  float* logits = (float*)d_out;

  for (int step = 0; step < 2; ++step) {
    hipMemsetAsync(mi, 0, (size_t)cells * HID * 4, stream);
    hipMemsetAsync(mo, 0, (size_t)cells * HID * 4, stream);
    // edge type 0 (intra)
    edge_mlp_kernel<<<eblocks, 256, 0, stream>>>(hb, esrc, edst, W0b, b0, W1b, b1, W2b, b2, W3b,
                                                 b3, mi, E);
    // edge type 1 (inter)
    edge_mlp_kernel<<<eblocks, 256, 0, stream>>>(hb, esrc + E, edst + E, W0b + 128 * 256,
                                                 b0 + 128, W1b + 128 * 128, b1 + 128,
                                                 W2b + 128 * 128, b2 + 128, W3b + 128 * 128,
                                                 b3 + 128, mo, E);
    // fused LSTM + score; updates h/rh (bf16) and rc (f32) in place
    lstm_kernel<<<lblocks, 256, 0, stream>>>(xb, rhb, mi, mo, Wcb, sw, hb, rhb, rc,
                                             logits + (size_t)step * cells);
  }
}
